// TemporalFGNNIwithInformer_75050258530315
// MI455X (gfx1250) — matrix operand order repaired and checked by reading, not compiled, not run on hardware
//
#include <hip/hip_runtime.h>
#include <cmath>

#define DM    512
#define NH    8
#define DKH   64
#define DFF   2048
#define NB    32
#define LEN_E 192
#define LEN_D 120
#define NNODE 128
#define PRE   24

typedef __attribute__((ext_vector_type(16))) _Float16 v16h;
typedef __attribute__((ext_vector_type(8)))  _Float16 v8h;
typedef __attribute__((ext_vector_type(8)))  float    v8f;

// ---------------- wave helpers (wave32) ----------------
__device__ __forceinline__ float wave_sum(float v){
#pragma unroll
  for (int m = 16; m > 0; m >>= 1) v += __shfl_xor(v, m, 32);
  return v;
}
__device__ __forceinline__ float wave_max(float v){
#pragma unroll
  for (int m = 16; m > 0; m >>= 1) v = fmaxf(v, __shfl_xor(v, m, 32));
  return v;
}
__device__ __forceinline__ unsigned hash3(unsigned a, unsigned b, unsigned c){
  unsigned x = a*0x9E3779B1u + b*0x85EBCA77u + c*0xC2B2AE3Du + 0x27D4EB2Fu;
  x ^= x >> 16; x *= 0x7FEB352Du; x ^= x >> 15; x *= 0x846CA68Bu; x ^= x >> 16;
  return x;
}

// ---------------- f32 -> f16 staging ----------------
__global__ void cvt_kernel(const float* __restrict__ in, _Float16* __restrict__ out, int n)
{
  int gid = blockIdx.x * blockDim.x + threadIdx.x;
  if (gid >= n) return;
  out[gid] = (_Float16)in[gid];
}

// ---------------- WMMA GEMM: C[M,N] = A[M,K] @ W[N,K]^T + bias ----------------
// A row-major [M,K] f16 (staged), W row-major [N,K] f16 (staged), f32 accumulate.
// Each wave computes a 16x64 strip: 4 WMMAs per K-step sharing one A fragment.
// Requires M%16==0, N%64==0, K%32==0 (all call sites satisfy this).
__global__ __launch_bounds__(256) void gemm_wmma_kernel(
    const _Float16* __restrict__ A, const _Float16* __restrict__ W,
    const float* __restrict__ bias, float* __restrict__ C,
    int M, int N, int K)
{
  const int lane = threadIdx.x & 31;
  const int ntn4 = N >> 6;
  const int tiles = (M >> 4) * ntn4;
  const int tile = blockIdx.x * 8 + (threadIdx.x >> 5);
  if (tile >= tiles) return;                    // uniform per wave -> EXEC all-ones for WMMA
  const int tm  = tile / ntn4;
  const int tn4 = tile - tm * ntn4;
  const int l15 = lane & 15;
  const int hi  = (lane >> 4) & 1;
  // A 16x32 f16 layout: lane<16: h[j]=K j ; lane>=16: K 16+j
  const _Float16* arow = A + (size_t)(tm*16 + l15) * K + hi*16;
  // B 32x16 f16 layout: lane<16: h[j]=K j (N=lane) ; lane>=16: K 16+j
  const _Float16* brow = W + (size_t)(tn4*64 + l15) * K + hi*16;
  const size_t bstride = (size_t)16 * K;        // next 16-column tile of the strip
  v8f acc0, acc1, acc2, acc3;
#pragma unroll
  for (int r = 0; r < 8; ++r) { acc0[r]=0.f; acc1[r]=0.f; acc2[r]=0.f; acc3[r]=0.f; }
#pragma unroll 2
  for (int k0 = 0; k0 < K; k0 += 32) {
    __builtin_prefetch(arow + k0 + 64, 0, 1);
    __builtin_prefetch(brow + k0 + 64, 0, 1);
    v8h a0 = *(const v8h*)(arow + k0);
    v8h a1 = *(const v8h*)(arow + k0 + 8);
    v16h av = __builtin_shufflevector(a0, a1, 0,1,2,3,4,5,6,7,8,9,10,11,12,13,14,15);
    v16h b0 = *(const v16h*)(brow + k0);
    v16h b1 = *(const v16h*)(brow + bstride   + k0);
    v16h b2 = *(const v16h*)(brow + bstride*2 + k0);
    v16h b3 = *(const v16h*)(brow + bstride*3 + k0);
    acc0 = __builtin_amdgcn_wmma_f32_16x16x32_f16(false, av, false, b0, (short)0, acc0, false, false);
    acc1 = __builtin_amdgcn_wmma_f32_16x16x32_f16(false, av, false, b1, (short)0, acc1, false, false);
    acc2 = __builtin_amdgcn_wmma_f32_16x16x32_f16(false, av, false, b2, (short)0, acc2, false, false);
    acc3 = __builtin_amdgcn_wmma_f32_16x16x32_f16(false, av, false, b3, (short)0, acc3, false, false);
  }
  const int mrow = tm*16 + hi*8;
#pragma unroll
  for (int t = 0; t < 4; ++t) {
    v8f acc = (t==0)?acc0:(t==1)?acc1:(t==2)?acc2:acc3;
    const int n = tn4*64 + t*16 + l15;
    const float bi = bias ? bias[n] : 0.f;
    float* crow = C + (size_t)mrow * N + n;
#pragma unroll
    for (int r = 0; r < 8; ++r) crow[(size_t)r * N] = acc[r] + bi;
  }
}

// ---------------- embedding: circular conv1d(k=3) + sinusoidal pos ----------------
__global__ void embed_kernel(const float* __restrict__ x, const float* __restrict__ W,
                             float* __restrict__ out, int L)
{
  int gid = blockIdx.x * blockDim.x + threadIdx.x;
  int total = NB * L * DM;
  if (gid >= total) return;
  int d  = gid & (DM - 1);
  int bl = gid >> 9;              // DM == 512
  int l  = bl % L, b = bl / L;
  const float* x0 = x + ((size_t)b*L + (l == 0   ? L-1 : l-1)) * NNODE;
  const float* x1 = x + ((size_t)b*L + l) * NNODE;
  const float* x2 = x + ((size_t)b*L + (l == L-1 ? 0   : l+1)) * NNODE;
  const float* wr = W + (size_t)d * (NNODE*3);
  float s = 0.f;
  for (int n = 0; n < NNODE; ++n)
    s += x0[n]*wr[n*3+0] + x1[n]*wr[n*3+1] + x2[n]*wr[n*3+2];
  int i = d >> 1;
  float dv  = expf(-(float)(2*i) * (9.210340371976184f / 512.0f)); // ln(10000)/d_model
  float ang = (float)l * dv;
  s += (d & 1) ? cosf(ang) : sinf(ang);
  out[gid] = s;
}

// build decoder input series: [x_enc[:, -96:], zeros(B, 24)]  -> [B, 120, 128]
__global__ void decbuild_kernel(const float* __restrict__ xe, float* __restrict__ out)
{
  int gid = blockIdx.x * blockDim.x + threadIdx.x;
  int total = NB * LEN_D * NNODE;
  if (gid >= total) return;
  int n = gid & (NNODE - 1);
  int l = (gid >> 7) % LEN_D;
  int b = gid / (LEN_D * NNODE);
  float v = 0.f;
  if (l < LEN_E/2) v = xe[((size_t)b*LEN_E + (LEN_E/2 + l)) * NNODE + n];
  out[gid] = v;
}

// ---------------- ProbSparse attention pieces ----------------
// M[b,h,l] = max_s(q.k_sample) - sum_s(q.k_sample)/Lk   (one wave per (b,h,l))
__global__ void probM_kernel(const float* __restrict__ q, const float* __restrict__ k,
                             float* __restrict__ Mout, int Lq, int Lk, int Up, unsigned keyid)
{
  int wid  = (blockIdx.x * blockDim.x + threadIdx.x) >> 5;
  int lane = threadIdx.x & 31;
  int total = NB * NH * Lq;
  if (wid >= total) return;
  int l  = wid % Lq;
  int bh = wid / Lq;
  int h = bh % NH, b = bh / NH;
  const float* qp = q + ((size_t)b*Lq + l) * DM + h*DKH;
  float dmax = -__builtin_inff();
  float dsum = 0.f;
  if (lane < Up) {
    int j = (int)(hash3(keyid, (unsigned)l, (unsigned)lane) % (unsigned)Lk);
    const float* kp = k + ((size_t)b*Lk + j) * DM + h*DKH;
    float acc = 0.f;
    for (int d = 0; d < DKH; ++d) acc += qp[d]*kp[d];
    dmax = acc; dsum = acc;
  }
  float mx = wave_max(dmax);
  float sm = wave_sum(dsum);
  if (lane == 0) Mout[wid] = mx - sm / (float)Lk;
}

// top-u selection per (b,h); ties -> smallest index (matches lax.top_k)
__global__ void topk_kernel(const float* __restrict__ Mv, int* __restrict__ top,
                            int Lq, int u)
{
  int t = blockIdx.x * blockDim.x + threadIdx.x;
  if (t >= NB*NH) return;
  const float* m = Mv + (size_t)t * Lq;
  unsigned long long taken[3] = {0ull, 0ull, 0ull};
  for (int j = 0; j < u; ++j) {
    float best = -__builtin_inff(); int bi = 0;
    for (int l = 0; l < Lq; ++l) {
      if ((taken[l >> 6] >> (l & 63)) & 1ull) continue;
      float val = m[l];
      if (val > best) { best = val; bi = l; }
    }
    taken[bi >> 6] |= 1ull << (bi & 63);
    top[(size_t)t * u + j] = bi;
  }
}

// context init: mean of V over Lk, broadcast to Lq rows (non-causal)
__global__ void ctx_mean_kernel(const float* __restrict__ v, float* __restrict__ ctx,
                                int Lq, int Lk)
{
  int gid = blockIdx.x * blockDim.x + threadIdx.x;
  if (gid >= NB*NH*DKH) return;
  int d  = gid & (DKH - 1);
  int bh = gid >> 6;
  int h = bh % NH, b = bh / NH;
  float s = 0.f;
  for (int kk = 0; kk < Lk; ++kk) s += v[((size_t)b*Lk + kk)*DM + h*DKH + d];
  s /= (float)Lk;
  for (int l = 0; l < Lq; ++l) ctx[((size_t)b*Lq + l)*DM + h*DKH + d] = s;
}

// context init: cumulative sum of V (causal; Lq == Lk)
__global__ void ctx_cumsum_kernel(const float* __restrict__ v, float* __restrict__ ctx,
                                  int Lq, int Lk)
{
  int gid = blockIdx.x * blockDim.x + threadIdx.x;
  if (gid >= NB*NH*DKH) return;
  int d  = gid & (DKH - 1);
  int bh = gid >> 6;
  int h = bh % NH, b = bh / NH;
  float run = 0.f;
  for (int l = 0; l < Lq; ++l) {
    run += v[((size_t)b*Lk + l)*DM + h*DKH + d];
    ctx[((size_t)b*Lq + l)*DM + h*DKH + d] = run;
  }
}

// per selected query row: scores vs all K, softmax, attn@V -> scatter into ctx
__global__ __launch_bounds__(32) void attn_upd_kernel(
    const float* __restrict__ q, const float* __restrict__ k, const float* __restrict__ v,
    const int* __restrict__ top, float* __restrict__ ctx,
    int Lq, int Lk, int u, int causal)
{
  __shared__ float sc[LEN_E];
  int ui = blockIdx.x % u;
  int bh = blockIdx.x / u;
  int h = bh % NH, b = bh / NH;
  int lane = threadIdx.x;
  int row = top[(size_t)bh * u + ui];
  const float* qp = q + ((size_t)b*Lq + row)*DM + h*DKH;
  float lmax = -__builtin_inff();
  for (int kk = lane; kk < Lk; kk += 32) {
    const float* kp = k + ((size_t)b*Lk + kk)*DM + h*DKH;
    float acc = 0.f;
    for (int d = 0; d < DKH; ++d) acc += qp[d]*kp[d];
    float s = acc * 0.125f;                    // 1/sqrt(64)
    if (causal && kk > row) s = -__builtin_inff();
    sc[kk] = s;
    lmax = fmaxf(lmax, s);
  }
  float m = wave_max(lmax);
  float lsum = 0.f;
  for (int kk = lane; kk < Lk; kk += 32) {
    float e = expf(sc[kk] - m);
    sc[kk] = e;
    lsum += e;
  }
  float inv = 1.f / wave_sum(lsum);
  __syncthreads();
  int d0 = lane * 2;
  float a0 = 0.f, a1 = 0.f;
  for (int kk = 0; kk < Lk; ++kk) {
    float w = sc[kk];
    const float* vp = v + ((size_t)b*Lk + kk)*DM + h*DKH;
    a0 += w * vp[d0];
    a1 += w * vp[d0 + 1];
  }
  float* cp = ctx + ((size_t)b*Lq + row)*DM + h*DKH;
  cp[d0]     = a0 * inv;
  cp[d0 + 1] = a1 * inv;
}

// ---------------- LayerNorm (optionally fused residual add), one wave per row ----------------
__global__ __launch_bounds__(256) void ln_kernel(
    const float* __restrict__ x, const float* __restrict__ r,
    const float* __restrict__ g, const float* __restrict__ bt,
    float* __restrict__ out, int rows)
{
  int wid  = (blockIdx.x * blockDim.x + threadIdx.x) >> 5;
  int lane = threadIdx.x & 31;
  if (wid >= rows) return;
  const float* xp = x + (size_t)wid * DM;
  const float* rp = r ? r + (size_t)wid * DM : nullptr;
  float vbuf[16];
  float s = 0.f;
#pragma unroll
  for (int i = 0; i < 16; ++i) {
    int idx = i*32 + lane;
    float t = xp[idx] + (rp ? rp[idx] : 0.f);
    vbuf[i] = t; s += t;
  }
  float mu = wave_sum(s) / (float)DM;
  float s2 = 0.f;
#pragma unroll
  for (int i = 0; i < 16; ++i) { float d = vbuf[i] - mu; s2 += d*d; }
  float inv = rsqrtf(wave_sum(s2) / (float)DM + 1e-5f);
  float* op = out + (size_t)wid * DM;
#pragma unroll
  for (int i = 0; i < 16; ++i) {
    int idx = i*32 + lane;
    op[idx] = (vbuf[i] - mu) * inv * g[idx] + bt[idx];
  }
}

// exact GELU
__global__ void gelu_kernel(float* __restrict__ p, int n)
{
  int gid = blockIdx.x * blockDim.x + threadIdx.x;
  if (gid >= n) return;
  float x = p[gid];
  p[gid] = 0.5f * x * (1.f + erff(x * 0.70710678118654752f));
}

// final slice: last PRE rows of projected decoder -> output [B, PRE, N]
__global__ void slice_kernel(const float* __restrict__ pin, float* __restrict__ out)
{
  int gid = blockIdx.x * blockDim.x + threadIdx.x;
  int total = NB * PRE * NNODE;
  if (gid >= total) return;
  int n = gid & (NNODE - 1);
  int t = (gid >> 7) % PRE;
  int b = gid / (PRE * NNODE);
  out[gid] = pin[((size_t)b * LEN_D + (LEN_D - PRE) + t) * NNODE + n];
}

// ================= host orchestration =================
static void gemm(hipStream_t s, const float* A, const float* W, const float* bias,
                 float* C, int M, int N, int K, _Float16* ah, _Float16* wh)
{
  { int n = M * K; cvt_kernel<<<(n + 255) / 256, 256, 0, s>>>(A, ah, n); }
  { int n = N * K; cvt_kernel<<<(n + 255) / 256, 256, 0, s>>>(W, wh, n); }
  int tiles = (M >> 4) * (N >> 6);
  gemm_wmma_kernel<<<(tiles + 7) / 8, 256, 0, s>>>(ah, wh, bias, C, M, N, K);
}
static void layer_norm(hipStream_t s, const float* x, const float* r,
                       const float* g, const float* b, float* out, int rows)
{
  ln_kernel<<<(rows * 32 + 255) / 256, 256, 0, s>>>(x, r, g, b, out, rows);
}
static int ufactor(int L) { int v = 5 * (int)ceil(log((double)L)); return v < L ? v : L; }

static void attention(hipStream_t s, const float* xq, int Lq, const float* xkv, int Lk,
                      void* const* d_in, int base, bool causal, unsigned keyid,
                      float* qb, float* kb, float* vb, float* ctxb,
                      float* Mb, int* topb, float* outb, _Float16* ah, _Float16* wh)
{
  const float* wq = (const float*)d_in[base+0]; const float* bq = (const float*)d_in[base+1];
  const float* wk = (const float*)d_in[base+2]; const float* bk = (const float*)d_in[base+3];
  const float* wv = (const float*)d_in[base+4]; const float* bv = (const float*)d_in[base+5];
  const float* wo = (const float*)d_in[base+6]; const float* bo = (const float*)d_in[base+7];
  int Mq = NB * Lq, Mk = NB * Lk;
  gemm(s, xq,  wq, bq, qb, Mq, DM, DM, ah, wh);
  gemm(s, xkv, wk, bk, kb, Mk, DM, DM, ah, wh);
  gemm(s, xkv, wv, bv, vb, Mk, DM, DM, ah, wh);
  int Up = ufactor(Lk), u = ufactor(Lq);
  int waves = NB * NH * Lq;
  probM_kernel<<<(waves * 32 + 255) / 256, 256, 0, s>>>(qb, kb, Mb, Lq, Lk, Up, keyid);
  topk_kernel<<<1, 256, 0, s>>>(Mb, topb, Lq, u);
  int th = NB * NH * DKH;
  if (causal) ctx_cumsum_kernel<<<(th + 255) / 256, 256, 0, s>>>(vb, ctxb, Lq, Lk);
  else        ctx_mean_kernel  <<<(th + 255) / 256, 256, 0, s>>>(vb, ctxb, Lq, Lk);
  attn_upd_kernel<<<NB * NH * u, 32, 0, s>>>(qb, kb, vb, topb, ctxb, Lq, Lk, u, causal ? 1 : 0);
  gemm(s, ctxb, wo, bo, outb, Mq, DM, DM, ah, wh);
}

extern "C" void kernel_launch(void* const* d_in, const int* in_sizes, int n_in,
                              void* d_out, int out_size, void* d_ws, size_t ws_size,
                              hipStream_t stream)
{
  (void)in_sizes; (void)n_in; (void)out_size; (void)ws_size;
  auto F = [&](int i) -> const float* { return (const float*)d_in[i]; };
  // param index map (dict insertion order):
  // 0:x_enc 1:enc_emb 2:dec_emb
  // enc layer i (base=3+16i): wq bq wk bk wv bv wo bo n1g n1b n2g n2b w1 b1 w2 b2
  // 51,52: enc_norm g,b
  // dec layer j (base=53+26j): sattn(8) cattn(8) n1g n1b n2g n2b n3g n3b w1 b1 w2 b2
  // 105,106: dec_norm g,b ; 107: proj_w ; 108: proj_b
  const float* x_enc = F(0);

  char* wsb = (char*)d_ws; size_t off = 0;
  auto allocb = [&](size_t nbytes) -> void* {
    off = (off + 255) & ~(size_t)255;
    void* p = (void*)(wsb + off);
    off += nbytes;
    return p;
  };
  auto alloc = [&](size_t nfloats) -> float* { return (float*)allocb(nfloats * sizeof(float)); };

  float* dec_in = alloc((size_t)NB * LEN_D * NNODE);
  float* encx   = alloc((size_t)NB * LEN_E * DM);
  float* decx   = alloc((size_t)NB * LEN_D * DM);
  float* qb     = alloc((size_t)NB * LEN_E * DM);
  float* kb     = alloc((size_t)NB * LEN_E * DM);
  float* vb     = alloc((size_t)NB * LEN_E * DM);
  float* ctxb   = alloc((size_t)NB * LEN_E * DM);
  float* t0     = alloc((size_t)NB * LEN_E * DM);
  float* hb     = alloc((size_t)NB * LEN_E * DFF);
  float* Mb     = alloc((size_t)NB * NH * LEN_E);
  int*   topb   = (int*)alloc((size_t)NB * NH * 32);
  float* pout   = alloc((size_t)NB * LEN_D * NNODE);
  _Float16* ah  = (_Float16*)allocb((size_t)NB * LEN_E * DFF * sizeof(_Float16)); // staged A (max M*K)
  _Float16* wh  = (_Float16*)allocb((size_t)DFF * DM * sizeof(_Float16));         // staged W (max N*K)

  // embeddings
  { int tot = NB*LEN_E*DM;    embed_kernel<<<(tot+255)/256,256,0,stream>>>(x_enc, F(1), encx, LEN_E); }
  { int tot = NB*LEN_D*NNODE; decbuild_kernel<<<(tot+255)/256,256,0,stream>>>(x_enc, dec_in); }
  { int tot = NB*LEN_D*DM;    embed_kernel<<<(tot+255)/256,256,0,stream>>>(dec_in, F(2), decx, LEN_D); }

  // encoder
  for (int i = 0; i < 3; ++i) {
    int base = 3 + i*16;
    attention(stream, encx, LEN_E, encx, LEN_E, d_in, base, false, (unsigned)i,
              qb, kb, vb, ctxb, Mb, topb, t0, ah, wh);
    layer_norm(stream, encx, t0, F(base+8), F(base+9), encx, NB*LEN_E);
    gemm(stream, encx, F(base+12), F(base+13), hb, NB*LEN_E, DFF, DM, ah, wh);
    { int n = NB*LEN_E*DFF; gelu_kernel<<<(n+255)/256,256,0,stream>>>(hb, n); }
    gemm(stream, hb, F(base+14), F(base+15), t0, NB*LEN_E, DM, DFF, ah, wh);
    layer_norm(stream, encx, t0, F(base+10), F(base+11), encx, NB*LEN_E);
  }
  layer_norm(stream, encx, nullptr, F(51), F(52), encx, NB*LEN_E);

  // decoder
  for (int j = 0; j < 2; ++j) {
    int base = 53 + j*26;
    attention(stream, decx, LEN_D, decx, LEN_D, d_in, base+0, true, 100u + (unsigned)j,
              qb, kb, vb, ctxb, Mb, topb, t0, ah, wh);
    layer_norm(stream, decx, t0, F(base+16), F(base+17), decx, NB*LEN_D);
    attention(stream, decx, LEN_D, encx, LEN_E, d_in, base+8, false, 150u + (unsigned)j,
              qb, kb, vb, ctxb, Mb, topb, t0, ah, wh);
    layer_norm(stream, decx, t0, F(base+18), F(base+19), decx, NB*LEN_D);
    gemm(stream, decx, F(base+22), F(base+23), hb, NB*LEN_D, DFF, DM, ah, wh);
    { int n = NB*LEN_D*DFF; gelu_kernel<<<(n+255)/256,256,0,stream>>>(hb, n); }
    gemm(stream, hb, F(base+24), F(base+25), t0, NB*LEN_D, DM, DFF, ah, wh);
    layer_norm(stream, decx, t0, F(base+20), F(base+21), decx, NB*LEN_D);
  }
  layer_norm(stream, decx, nullptr, F(105), F(106), decx, NB*LEN_D);

  // projection + slice
  gemm(stream, decx, F(107), F(108), pout, NB*LEN_D, NNODE, DM, ah, wh);
  { int tot = NB*PRE*NNODE; slice_kernel<<<(tot+255)/256,256,0,stream>>>(pout, (float*)d_out); }
}
